// InfoNCE_26834955666078
// MI455X (gfx1250) — compile-verified, matrix operand order checked by reading, not run on hardware
//
#include <hip/hip_runtime.h>
#include <hip/hip_bf16.h>
#include <math.h>

#define D        1024
#define P_PAIRS  2048
#define N_PAIRS  65536
#define U_SIZE   4096
#define EPS      1e-8f
#define BM       32          // rows per workgroup (2 M-tiles of 16)
#define WAVES    8
#define TPB      256
// LDS: Hs(64K) + Ht(64K) + Ps(64K) + Pt(64K) + red(3K)
#define SMEM_BYTES 265216

typedef __attribute__((ext_vector_type(16))) __bf16 v16bf;
typedef __attribute__((ext_vector_type(8)))  float  v8f;

struct __align__(16) U4 { unsigned int a[4]; };

union BfVec {
    v16bf          v;
    U4             q[2];
    unsigned short us[16];
};

__device__ __forceinline__ unsigned short f2bf(float f) {
    unsigned int u = __float_as_uint(f);
    u += 0x7FFFu + ((u >> 16) & 1u);      // round-to-nearest-even
    return (unsigned short)(u >> 16);
}
__device__ __forceinline__ float bf2f(unsigned short h) {
    return __uint_as_float(((unsigned int)h) << 16);
}
__device__ __forceinline__ float gelu_exact(float x) {
    return 0.5f * x * (1.0f + erff(x * 0.70710678118654752f));
}

// ---------------------------------------------------------------------------
// Transpose W (f32, [k][n]) -> WT (bf16, [n][k]) so WMMA B loads mirror A loads.
// ---------------------------------------------------------------------------
__global__ void __launch_bounds__(256) wt_convert(const float* __restrict__ W,
                                                  unsigned short* __restrict__ WT) {
    __shared__ float tile[32][33];
    const int bx = blockIdx.x, by = blockIdx.y;
    const int tx = threadIdx.x, ty = threadIdx.y;
    #pragma unroll
    for (int i = 0; i < 32; i += 8)
        tile[ty + i][tx] = W[(size_t)(by * 32 + ty + i) * D + bx * 32 + tx];
    __syncthreads();
    #pragma unroll
    for (int i = 0; i < 32; i += 8)
        WT[(size_t)(bx * 32 + ty + i) * D + by * 32 + tx] = f2bf(tile[tx][ty + i]);
}

// ---------------------------------------------------------------------------
// Fused head pair + cosine for a 32-row block.  Each wave owns 8 N-tiles; each
// B register load feeds TWO v_wmma (M-tile 0: rows 0-15, M-tile 1: rows 16-31),
// halving L2 weight traffic per output row.
// ---------------------------------------------------------------------------
__global__ void __launch_bounds__(256) fused_proj_cos(
    const float* __restrict__ Xs, const float* __restrict__ Xt,
    const unsigned short* __restrict__ W1s, const float* __restrict__ b1s,
    const unsigned short* __restrict__ W2s, const float* __restrict__ b2s,
    const unsigned short* __restrict__ W1t, const float* __restrict__ b1t,
    const unsigned short* __restrict__ W2t, const float* __restrict__ b2t,
    const float* __restrict__ tempPtr,
    float* __restrict__ sims)
{
    extern __shared__ char smem[];
    unsigned short* Hs = (unsigned short*)smem;   // X_s bf16, later H_s bf16
    unsigned short* Ht = Hs + BM * D;             // X_t bf16, later H_t bf16
    unsigned short* Ps = Ht + BM * D;             // P_s bf16
    unsigned short* Pt = Ps + BM * D;             // P_t bf16
    float* red = (float*)(smem + 4 * BM * D * 2); // 32 rows x 8 waves x 3 f32

    const int tid  = threadIdx.x;
    const int lane = tid & 31;
    const int wave = tid >> 5;
    const int hl   = lane >> 4;       // half-wave (K-chunk phase)
    const int lm   = lane & 15;       // A row / B,C column within tile
    const int row0 = blockIdx.x * BM;

    // Phase 0: stage X rows (f32 -> bf16) into LDS, b128 reads / b64 writes.
    {
        const float4* xs4 = (const float4*)(Xs + (size_t)row0 * D);
        const float4* xt4 = (const float4*)(Xt + (size_t)row0 * D);
        uint2* hs2 = (uint2*)Hs;
        uint2* ht2 = (uint2*)Ht;
        for (int i = tid; i < BM * D / 4; i += TPB) {
            float4 f = xs4[i];
            uint2 u;
            u.x = (unsigned)f2bf(f.x) | ((unsigned)f2bf(f.y) << 16);
            u.y = (unsigned)f2bf(f.z) | ((unsigned)f2bf(f.w) << 16);
            hs2[i] = u;
            float4 g = xt4[i];
            uint2 v;
            v.x = (unsigned)f2bf(g.x) | ((unsigned)f2bf(g.y) << 16);
            v.y = (unsigned)f2bf(g.z) | ((unsigned)f2bf(g.w) << 16);
            ht2[i] = v;
        }
    }
    __syncthreads();

    // Phase 1: P = X @ W1 + b1 (both streams), P stored bf16.
    for (int s = 0; s < 2; ++s) {
        const unsigned short* Ab = s ? Ht : Hs;
        const unsigned short* W1 = s ? W1t : W1s;
        const float*          b1 = s ? b1t : b1s;
        unsigned short*       Pb = s ? Pt  : Ps;
        const unsigned short* arow0 = Ab + lm * D + hl * 8;
        const unsigned short* arow1 = Ab + (16 + lm) * D + hl * 8;
        for (int t = 0; t < 8; ++t) {
            const int colg = (wave * 8 + t) * 16 + lm;
            const unsigned short* brow = W1 + (size_t)colg * D + hl * 8;
            v8f acc0 = {}, acc1 = {};
            #pragma unroll 2
            for (int k0 = 0; k0 < D; k0 += 32) {
                BfVec b, a0, a1;
                b.q[0]  = *(const U4*)(brow + k0);
                b.q[1]  = *(const U4*)(brow + k0 + 16);
                a0.q[0] = *(const U4*)(arow0 + k0);
                a0.q[1] = *(const U4*)(arow0 + k0 + 16);
                a1.q[0] = *(const U4*)(arow1 + k0);
                a1.q[1] = *(const U4*)(arow1 + k0 + 16);
                acc0 = __builtin_amdgcn_wmma_f32_16x16x32_bf16(
                         false, a0.v, false, b.v, (short)0, acc0, false, false);
                acc1 = __builtin_amdgcn_wmma_f32_16x16x32_bf16(
                         false, a1.v, false, b.v, (short)0, acc1, false, false);
            }
            const float bias = b1[colg];
            #pragma unroll
            for (int r = 0; r < 8; ++r) {
                Pb[(r + hl * 8) * D + colg]        = f2bf(acc0[r] + bias);
                Pb[(16 + r + hl * 8) * D + colg]   = f2bf(acc1[r] + bias);
            }
        }
    }
    __syncthreads();

    // Phase 1.5: H = bf16(gelu(P)); X staging regions are dead, reuse them.
    {
        const uint2* ps2 = (const uint2*)Ps;
        const uint2* pt2 = (const uint2*)Pt;
        uint2* hs2 = (uint2*)Hs;
        uint2* ht2 = (uint2*)Ht;
        for (int i = tid; i < BM * D / 4; i += TPB) {
            uint2 p = ps2[i], u;
            u.x = (unsigned)f2bf(gelu_exact(bf2f((unsigned short)(p.x & 0xFFFF))))
                | ((unsigned)f2bf(gelu_exact(bf2f((unsigned short)(p.x >> 16)))) << 16);
            u.y = (unsigned)f2bf(gelu_exact(bf2f((unsigned short)(p.y & 0xFFFF))))
                | ((unsigned)f2bf(gelu_exact(bf2f((unsigned short)(p.y >> 16)))) << 16);
            hs2[i] = u;
            uint2 q = pt2[i], v;
            v.x = (unsigned)f2bf(gelu_exact(bf2f((unsigned short)(q.x & 0xFFFF))))
                | ((unsigned)f2bf(gelu_exact(bf2f((unsigned short)(q.x >> 16)))) << 16);
            v.y = (unsigned)f2bf(gelu_exact(bf2f((unsigned short)(q.y & 0xFFFF))))
                | ((unsigned)f2bf(gelu_exact(bf2f((unsigned short)(q.y >> 16)))) << 16);
            ht2[i] = v;
        }
    }
    __syncthreads();

    // Phase 2: Z = H @ W2 + b2 + P (both streams) fused with cosine partials.
    float dotA[2][8] = {}, nsA[2][8] = {}, ntA[2][8] = {};
    const unsigned short* hs0 = Hs + lm * D + hl * 8;
    const unsigned short* hs1 = Hs + (16 + lm) * D + hl * 8;
    const unsigned short* ht0 = Ht + lm * D + hl * 8;
    const unsigned short* ht1 = Ht + (16 + lm) * D + hl * 8;
    for (int t = 0; t < 8; ++t) {
        const int colg = (wave * 8 + t) * 16 + lm;
        const unsigned short* bsrow = W2s + (size_t)colg * D + hl * 8;
        const unsigned short* btrow = W2t + (size_t)colg * D + hl * 8;
        v8f aS0 = {}, aS1 = {}, aT0 = {}, aT1 = {};
        #pragma unroll 2
        for (int k0 = 0; k0 < D; k0 += 32) {
            BfVec b, x0, x1;
            b.q[0]  = *(const U4*)(bsrow + k0);
            b.q[1]  = *(const U4*)(bsrow + k0 + 16);
            x0.q[0] = *(const U4*)(hs0 + k0);
            x0.q[1] = *(const U4*)(hs0 + k0 + 16);
            x1.q[0] = *(const U4*)(hs1 + k0);
            x1.q[1] = *(const U4*)(hs1 + k0 + 16);
            aS0 = __builtin_amdgcn_wmma_f32_16x16x32_bf16(
                    false, x0.v, false, b.v, (short)0, aS0, false, false);
            aS1 = __builtin_amdgcn_wmma_f32_16x16x32_bf16(
                    false, x1.v, false, b.v, (short)0, aS1, false, false);
            b.q[0]  = *(const U4*)(btrow + k0);
            b.q[1]  = *(const U4*)(btrow + k0 + 16);
            x0.q[0] = *(const U4*)(ht0 + k0);
            x0.q[1] = *(const U4*)(ht0 + k0 + 16);
            x1.q[0] = *(const U4*)(ht1 + k0);
            x1.q[1] = *(const U4*)(ht1 + k0 + 16);
            aT0 = __builtin_amdgcn_wmma_f32_16x16x32_bf16(
                    false, x0.v, false, b.v, (short)0, aT0, false, false);
            aT1 = __builtin_amdgcn_wmma_f32_16x16x32_bf16(
                    false, x1.v, false, b.v, (short)0, aT1, false, false);
        }
        const float biasS = b2s[colg], biasT = b2t[colg];
        #pragma unroll
        for (int r = 0; r < 8; ++r) {
            const int r0 = r + hl * 8;
            float zs = aS0[r] + biasS + bf2f(Ps[r0 * D + colg]);
            float zt = aT0[r] + biasT + bf2f(Pt[r0 * D + colg]);
            dotA[0][r] = fmaf(zs, zt, dotA[0][r]);
            nsA[0][r]  = fmaf(zs, zs, nsA[0][r]);
            ntA[0][r]  = fmaf(zt, zt, ntA[0][r]);
            const int r1 = 16 + r0;
            zs = aS1[r] + biasS + bf2f(Ps[r1 * D + colg]);
            zt = aT1[r] + biasT + bf2f(Pt[r1 * D + colg]);
            dotA[1][r] = fmaf(zs, zt, dotA[1][r]);
            nsA[1][r]  = fmaf(zs, zs, nsA[1][r]);
            ntA[1][r]  = fmaf(zt, zt, ntA[1][r]);
        }
    }

    // Cross-lane (16 columns per half-wave) then cross-wave reduction.
    #pragma unroll
    for (int m = 0; m < 2; ++m) {
        #pragma unroll
        for (int r = 0; r < 8; ++r) {
            float d = dotA[m][r], a = nsA[m][r], b = ntA[m][r];
            for (int off = 8; off >= 1; off >>= 1) {
                d += __shfl_xor(d, off, 32);
                a += __shfl_xor(a, off, 32);
                b += __shfl_xor(b, off, 32);
            }
            if (lm == 0) {
                const int rr = m * 16 + r + hl * 8;
                red[(rr * WAVES + wave) * 3 + 0] = d;
                red[(rr * WAVES + wave) * 3 + 1] = a;
                red[(rr * WAVES + wave) * 3 + 2] = b;
            }
        }
    }
    __syncthreads();
    if (tid < BM) {
        float d = 0.f, a = 0.f, b = 0.f;
        #pragma unroll
        for (int w = 0; w < WAVES; ++w) {
            d += red[(tid * WAVES + w) * 3 + 0];
            a += red[(tid * WAVES + w) * 3 + 1];
            b += red[(tid * WAVES + w) * 3 + 2];
        }
        const float temp = fminf(fmaxf(tempPtr[0], 0.001f), 0.5f);
        sims[row0 + tid] = d / (fmaxf(sqrtf(a), EPS) * fmaxf(sqrtf(b), EPS) * temp);
    }
}

// ---------------------------------------------------------------------------
// Epilogue: histograms, stable weighted logsumexp, loss.
// ---------------------------------------------------------------------------
__global__ void hist_zero(float* hs, float* ht) {
    const int i = blockIdx.x * blockDim.x + threadIdx.x;
    if (i < U_SIZE) { hs[i] = 0.f; ht[i] = 0.f; }
}

__global__ void hist_build(const int* __restrict__ pis, const int* __restrict__ pit,
                           float* hs, float* ht) {
    const int i = blockIdx.x * blockDim.x + threadIdx.x;
    if (i < P_PAIRS) {                 // +1.0 adds are exact integers: order-independent
        atomicAdd(&hs[pis[i]], 1.0f);
        atomicAdd(&ht[pit[i]], 1.0f);
    }
}

__global__ void __launch_bounds__(256) max_partial(const float* __restrict__ sims,
                                                   float* __restrict__ part, int n) {
    __shared__ float sm[256];
    float m = -INFINITY;
    for (int i = blockIdx.x * 256 + threadIdx.x; i < n; i += gridDim.x * 256)
        m = fmaxf(m, sims[i]);
    sm[threadIdx.x] = m;
    __syncthreads();
    for (int o = 128; o >= 1; o >>= 1) {
        if (threadIdx.x < o) sm[threadIdx.x] = fmaxf(sm[threadIdx.x], sm[threadIdx.x + o]);
        __syncthreads();
    }
    if (threadIdx.x == 0) part[blockIdx.x] = sm[0];
}

__global__ void __launch_bounds__(256) final_loss(
    const float* __restrict__ sims, const float* __restrict__ partMax, int nPart,
    const float* __restrict__ hs, const float* __restrict__ ht,
    const int* __restrict__ nis, const int* __restrict__ nit,
    float* __restrict__ out)
{
    __shared__ float  smx[256];
    __shared__ double swd[256];
    __shared__ double spd[256];
    __shared__ float  gmax;

    float m = -INFINITY;
    for (int i = threadIdx.x; i < nPart; i += 256) m = fmaxf(m, partMax[i]);
    smx[threadIdx.x] = m;
    __syncthreads();
    for (int o = 128; o >= 1; o >>= 1) {
        if (threadIdx.x < o) smx[threadIdx.x] = fmaxf(smx[threadIdx.x], smx[threadIdx.x + o]);
        __syncthreads();
    }
    if (threadIdx.x == 0) gmax = smx[0];
    __syncthreads();
    const float mM = gmax;

    double accW = 0.0, accP = 0.0;     // fixed-order, deterministic
    for (int i = threadIdx.x; i < P_PAIRS + N_PAIRS; i += 256) {
        const float v = sims[i];
        float w;
        if (i < P_PAIRS) { w = 1.0f; accP += (double)v; }
        else {
            const int j = i - P_PAIRS;
            w = hs[nis[j]] + ht[nit[j]];
        }
        accW += (double)(w * __expf(v - mM));
    }
    swd[threadIdx.x] = accW;
    spd[threadIdx.x] = accP;
    __syncthreads();
    for (int o = 128; o >= 1; o >>= 1) {
        if (threadIdx.x < o) {
            swd[threadIdx.x] += swd[threadIdx.x + o];
            spd[threadIdx.x] += spd[threadIdx.x + o];
        }
        __syncthreads();
    }
    if (threadIdx.x == 0) {
        const double lse = (double)mM + log(swd[0]);
        out[0] = (float)(lse - spd[0] / (double)P_PAIRS);   // (loss_st+loss_st)/2
    }
}

// ---------------------------------------------------------------------------
extern "C" void kernel_launch(void* const* d_in, const int* in_sizes, int n_in,
                              void* d_out, int out_size, void* d_ws, size_t ws_size,
                              hipStream_t stream) {
    (void)in_sizes; (void)n_in; (void)out_size; (void)ws_size;
    const float* pos_seq    = (const float*)d_in[0];
    const float* pos_text   = (const float*)d_in[1];
    const float* neg_seq    = (const float*)d_in[2];
    const float* neg_text   = (const float*)d_in[3];
    const int*   pos_inds_s = (const int*)  d_in[4];
    const int*   pos_inds_t = (const int*)  d_in[5];
    const int*   neg_inds_s = (const int*)  d_in[6];
    const int*   neg_inds_t = (const int*)  d_in[7];
    const float* temperature= (const float*)d_in[8];
    const float* Wp1 = (const float*)d_in[9];
    const float* bp1 = (const float*)d_in[10];
    const float* Wp2 = (const float*)d_in[11];
    const float* bp2 = (const float*)d_in[12];
    const float* Wr1 = (const float*)d_in[13];
    const float* br1 = (const float*)d_in[14];
    const float* Wr2 = (const float*)d_in[15];
    const float* br2 = (const float*)d_in[16];

    char* ws = (char*)d_ws;
    const size_t WSZ = (size_t)D * D * sizeof(unsigned short);   // 2 MB each
    unsigned short* WTp1 = (unsigned short*)(ws + 0 * WSZ);
    unsigned short* WTp2 = (unsigned short*)(ws + 1 * WSZ);
    unsigned short* WTr1 = (unsigned short*)(ws + 2 * WSZ);
    unsigned short* WTr2 = (unsigned short*)(ws + 3 * WSZ);
    float* sims    = (float*)(ws + 4 * WSZ);
    float* hist_s  = (float*)(ws + 4 * WSZ + 270336);
    float* hist_t  = hist_s + U_SIZE;
    float* partMax = hist_t + U_SIZE;

    const dim3 tb(32, 8), tg(D / 32, D / 32);
    wt_convert<<<tg, tb, 0, stream>>>(Wp1, WTp1);
    wt_convert<<<tg, tb, 0, stream>>>(Wp2, WTp2);
    wt_convert<<<tg, tb, 0, stream>>>(Wr1, WTr1);
    wt_convert<<<tg, tb, 0, stream>>>(Wr2, WTr2);

    (void)hipFuncSetAttribute((const void*)fused_proj_cos,
                              hipFuncAttributeMaxDynamicSharedMemorySize, SMEM_BYTES);
    fused_proj_cos<<<P_PAIRS / BM, TPB, SMEM_BYTES, stream>>>(
        pos_seq, pos_text, WTp1, bp1, WTp2, bp2, WTr1, br1, WTr2, br2,
        temperature, sims);
    fused_proj_cos<<<N_PAIRS / BM, TPB, SMEM_BYTES, stream>>>(
        neg_seq, neg_text, WTp1, bp1, WTp2, bp2, WTr1, br1, WTr2, br2,
        temperature, sims + P_PAIRS);

    hist_zero<<<(U_SIZE + 255) / 256, 256, 0, stream>>>(hist_s, hist_t);
    hist_build<<<(P_PAIRS + 255) / 256, 256, 0, stream>>>(pos_inds_s, pos_inds_t,
                                                          hist_s, hist_t);
    const int NT = P_PAIRS + N_PAIRS;
    const int MB = 264;
    max_partial<<<MB, 256, 0, stream>>>(sims, partMax, NT);
    final_loss<<<1, 256, 0, stream>>>(sims, partMax, MB, hist_s, hist_t,
                                      neg_inds_s, neg_inds_t, (float*)d_out);
}
